// MultiHeadAttention_36189394436198
// MI455X (gfx1250) — matrix operand order repaired and checked by reading, not compiled, not run on hardware
//
#include <hip/hip_runtime.h>

#define DEV __device__ __forceinline__

typedef __attribute__((ext_vector_type(16))) __bf16 v16bf;
typedef __attribute__((ext_vector_type(8)))  __bf16 v8bf;
typedef __attribute__((ext_vector_type(4)))  __bf16 v4bf;
typedef __attribute__((ext_vector_type(8)))  float  v8f;
typedef __attribute__((ext_vector_type(4)))  unsigned int v4u;
typedef __attribute__((ext_vector_type(8)))  int v8i;
typedef __attribute__((ext_vector_type(4)))  int v4i;

constexpr int B_ = 2, S_ = 2048, D_ = 1024, H_ = 16, DH_ = 64;
constexpr unsigned kStripBytes = 16u * S_ * 4u;   // 128 KB fp32 score strip

DEV v8f wmma_bf16(v16bf a, v16bf b, v8f c) {
  return __builtin_amdgcn_wmma_f32_16x16x32_bf16(
      /*neg_a=*/false, a, /*neg_b=*/false, b,
      /*c_mod=*/(short)0, c, /*reuse_a=*/false, /*reuse_b=*/false);
}

DEV v8f zero8() {
  v8f c;
#pragma unroll
  for (int i = 0; i < 8; ++i) c[i] = 0.0f;
  return c;
}

// A fragment: 16x32 (MxK), row-major bf16, leading dim ld.
DEV v16bf frag_a_bf16(const __bf16* base, int ld, int row0, int k0, int lane) {
  const int lo = lane & 15, hf = lane >> 4;
  const __bf16* p = base + (size_t)(row0 + lo) * ld + k0 + 8 * hf;
  v8bf x0 = *(const v8bf*)p;
  v8bf x1 = *(const v8bf*)(p + 16);
  v16bf a;
#pragma unroll
  for (int i = 0; i < 8; ++i) { a[i] = x0[i]; a[i + 8] = x1[i]; }
  return a;
}

// B fragment: 32x16 (KxN) with B[k][n] = R[n][k], R row-major bf16.
DEV v16bf frag_bt_bf16(const __bf16* R, int ld, int n0, int k0, int lane) {
  const int n = lane & 15, hf = lane >> 4;
  const __bf16* p = R + (size_t)(n0 + n) * ld + k0 + 8 * hf;
  v8bf x0 = *(const v8bf*)p;
  v8bf x1 = *(const v8bf*)(p + 16);
  v16bf b;
#pragma unroll
  for (int i = 0; i < 8; ++i) { b[i] = x0[i]; b[i + 8] = x1[i]; }
  return b;
}

// Tensor Data Mover: DMA a 16 x 2048 int32 tile (row stride S_) into LDS at
// byte offset `lds_off`, fully async (TENSORcnt). D# per CDNA5 ISA.
DEV void tdm_load_mask_strip(const int* gsrc, unsigned lds_off) {
  const unsigned long long ga = (unsigned long long)(uintptr_t)gsrc;
  v4u g0;
  g0[0] = 1u;                                   // count=1, user mode
  g0[1] = lds_off;                              // lds_addr (bytes)
  g0[2] = (unsigned)(ga & 0xffffffffu);         // global_addr[31:0]
  g0[3] = (unsigned)(ga >> 32) | 0x80000000u;   // global_addr[56:32] | type=2
  v8i g1;
  g1[0] = 0x20000;                              // data_size=4B (code 2)
  g1[1] = (int)(2048u << 16);                   // tensor_dim0 = 2048 (lo16)
  g1[2] = (int)(16u << 16);                     // dim0 hi=0 | tensor_dim1 = 16
  g1[3] = (int)(2048u << 16);                   // dim1 hi=0 | tile_dim0 = 2048
  g1[4] = 16;                                   // tile_dim1 = 16, tile_dim2 = 0
  g1[5] = 2048;                                 // tensor_dim0_stride = 2048
  g1[6] = 0;
  g1[7] = 0;
  v4i gz; gz[0] = 0; gz[1] = 0; gz[2] = 0; gz[3] = 0;   // 2D tensor: groups 2/3 unused
#if defined(__clang_major__) && __clang_major__ >= 23
  v8i gz8;
#pragma unroll
  for (int i = 0; i < 8; ++i) gz8[i] = 0;
  __builtin_amdgcn_tensor_load_to_lds(g0, g1, gz, gz, gz8, 0);
#else
  __builtin_amdgcn_tensor_load_to_lds(g0, g1, gz, gz, 0);
#endif
}

// ---------------- fp32 -> bf16 pre-convert (activations + weights) ----------
__global__ void __launch_bounds__(256) cvt_bf16_kernel(
    const float* __restrict__ in, __bf16* __restrict__ out, int n) {
  const int stride = gridDim.x * 256 * 4;
  for (int i = (blockIdx.x * 256 + threadIdx.x) * 4; i < n; i += stride) {
    const float4 v = *(const float4*)(in + i);
    v4bf o;
    o[0] = (__bf16)v.x; o[1] = (__bf16)v.y; o[2] = (__bf16)v.z; o[3] = (__bf16)v.w;
    *(v4bf*)(out + i) = o;
  }
}

// ---------------- Projection GEMM: Y(4096x1024) = X @ W^T + b ---------------
// One wave owns a 64x64 block (4x4 WMMA tiles): 16 WMMAs per 8 fragment loads.
template <int TRANS>
__global__ void __launch_bounds__(256) proj_kernel(
    const __bf16* __restrict__ X, const __bf16* __restrict__ W,
    const float* __restrict__ bias, __bf16* __restrict__ out) {
  const int wave = threadIdx.x >> 5, lane = threadIdx.x & 31;
  const int wtile = blockIdx.x * 8 + wave;      // 64 m-blocks x 16 n-blocks
  const int m0 = (wtile >> 4) * 64;
  const int n0 = (wtile & 15) * 64;
  v8f acc[4][4];
#pragma unroll
  for (int i = 0; i < 4; ++i)
#pragma unroll
    for (int j = 0; j < 4; ++j) acc[i][j] = zero8();

  for (int k0 = 0; k0 < D_; k0 += 32) {
    v16bf a[4], b[4];
#pragma unroll
    for (int i = 0; i < 4; ++i) a[i] = frag_a_bf16(X, D_, m0 + 16 * i, k0, lane);
#pragma unroll
    for (int j = 0; j < 4; ++j) b[j] = frag_bt_bf16(W, D_, n0 + 16 * j, k0, lane);
#pragma unroll
    for (int i = 0; i < 4; ++i)
#pragma unroll
      for (int j = 0; j < 4; ++j) acc[i][j] = wmma_bf16(a[i], b[j], acc[i][j]);
  }

  const int ln = lane & 15, hf = lane >> 4;
#pragma unroll
  for (int j = 0; j < 4; ++j) {
    const int n = n0 + 16 * j + ln;
    const float bv = bias[n];
    const int h = n >> 6, dh = n & (DH_ - 1);
#pragma unroll
    for (int i = 0; i < 4; ++i) {
#pragma unroll
      for (int r = 0; r < 8; ++r) {
        const int m = m0 + 16 * i + r + 8 * hf;
        const int bb = m >> 11, s = m & (S_ - 1);
        const size_t idx = TRANS
            ? ((((size_t)bb * H_ + h) * DH_ + dh) * S_ + s)
            : ((((size_t)bb * H_ + h) * S_ + s) * DH_ + dh);
        out[idx] = (__bf16)(acc[i][j][r] + bv);
      }
    }
  }
}

// ---- Fused attention, one workgroup (8 waves) per (bh, 16-query strip).
// LDS: 128 KB fp32 score strip + 128 KB TDM-staged mask strip (256 KB of the
// WGP's 320 KB). Mask DMA (TENSORcnt) overlaps the QK^T WMMA phase.
__global__ void __launch_bounds__(256) attn_fused_kernel(
    const __bf16* __restrict__ Qbf, const __bf16* __restrict__ Kbf,
    const __bf16* __restrict__ Vt, const int* __restrict__ mask,
    float* __restrict__ attn, __bf16* __restrict__ Xctx) {
  extern __shared__ float sP[];                 // [16*S_] scores, then mask strip
  int* mstrip = (int*)(sP + 16 * S_);           // 16 x 2048 int32 (TDM dest)
  __bf16* sPb = (__bf16*)(sP + 16 * S_);        // bf16 probs (reuses mask region)
  const int wave = threadIdx.x >> 5, lane = threadIdx.x & 31;
  const int bh = blockIdx.x >> 7;               // 128 q-strips per (b,h)
  const int q0 = (blockIdx.x & 127) << 4;
  const int bb = bh >> 4, h = bh & 15;
  const int hf = lane >> 4, ln = lane & 15;
  const __bf16* Qb = Qbf + (size_t)bh * S_ * DH_;
  const __bf16* Kb = Kbf + (size_t)bh * S_ * DH_;

  // Phase 0: wave 0 kicks off the async mask-strip DMA (runs under the WMMAs).
  if (wave == 0)
    tdm_load_mask_strip(mask + ((size_t)bb * S_ + q0) * S_, kStripBytes);

  // Q strip A-fragments (K=0..31, 32..63), reused across all 128 key tiles.
  v16bf aq0 = frag_a_bf16(Qb, DH_, q0, 0, lane);
  v16bf aq1 = frag_a_bf16(Qb, DH_, q0, 32, lane);

  // Phase 1: raw scaled scores into the LDS strip.
  for (int kt = wave; kt < (S_ >> 4); kt += 8) {
    const int n0 = kt << 4;
    __builtin_prefetch(Kb + (size_t)(n0 + 128 + ln) * DH_, 0, 0);
    v8f c = zero8();
    c = wmma_bf16(aq0, frag_bt_bf16(Kb, DH_, n0, 0, lane), c);
    c = wmma_bf16(aq1, frag_bt_bf16(Kb, DH_, n0, 32, lane), c);
    const int n = n0 + ln;
#pragma unroll
    for (int r = 0; r < 8; ++r)
      sP[(r + 8 * hf) * S_ + n] = c[r] * 0.125f;     // 1/sqrt(DH)
  }
  if (wave == 0) __builtin_amdgcn_s_wait_tensorcnt(0);
  __syncthreads();

  // Phase 1b: branchless 4-wide mask apply (ds_load_b128 / ds_store_b128).
  for (int i4 = threadIdx.x * 4; i4 < 16 * S_; i4 += 256 * 4) {
    const v4i m = *(const v4i*)(mstrip + i4);
    float4 v = *(const float4*)(sP + i4);
    v.x = (m[0] == 0) ? -1.0e9f : v.x;
    v.y = (m[1] == 0) ? -1.0e9f : v.y;
    v.z = (m[2] == 0) ? -1.0e9f : v.z;
    v.w = (m[3] == 0) ? -1.0e9f : v.w;
    *(float4*)(sP + i4) = v;
  }
  __syncthreads();

  // Phase 2: softmax; one wave owns 2 rows (wave32 shuffle reductions).
#pragma unroll
  for (int rr = 0; rr < 2; ++rr) {
    const int row = wave * 2 + rr;
    float* p = sP + row * S_;
    float m = -3.0e38f;
    for (int i = lane; i < S_; i += 32) m = fmaxf(m, p[i]);
#pragma unroll
    for (int off = 16; off > 0; off >>= 1) m = fmaxf(m, __shfl_xor(m, off, 32));
    float sum = 0.0f;
    for (int i = lane; i < S_; i += 32) { float e = __expf(p[i] - m); p[i] = e; sum += e; }
#pragma unroll
    for (int off = 16; off > 0; off >>= 1) sum += __shfl_xor(sum, off, 32);
    const float inv = 1.0f / sum;
    for (int i = lane; i < S_; i += 32) p[i] *= inv;
  }
  __syncthreads();

  // Phase 3: the attention strip is one contiguous 128 KB HBM block; write it
  // with float4 stores, fused with 4-wide bf16 packing into LDS (mask region
  // is dead after phase 1b -> reuse for bf16 probs).
  {
    float* arow = attn + ((size_t)bh * S_ + q0) * S_;
    for (int i4 = threadIdx.x * 4; i4 < 16 * S_; i4 += 256 * 4) {
      const float4 v = *(const float4*)(sP + i4);
      *(float4*)(arow + i4) = v;
      v4bf o;
      o[0] = (__bf16)v.x; o[1] = (__bf16)v.y; o[2] = (__bf16)v.z; o[3] = (__bf16)v.w;
      *(v4bf*)(sPb + i4) = o;
    }
  }
  __syncthreads();

  // Phase 4: PV from bf16 LDS probabilities. Waves 0..3 own the 4 d-tiles.
  if (wave < 4) {
    const int d0 = wave << 4;
    const __bf16* Vb = Vt + (size_t)bh * DH_ * S_;
    v8f c = zero8();
    for (int k0 = 0; k0 < S_; k0 += 32) {
      v16bf a = frag_a_bf16(sPb, S_, 0, k0, lane);    // 16B ds loads, no cvt
      v16bf b = frag_bt_bf16(Vb, S_, d0, k0, lane);   // contiguous (V transposed)
      c = wmma_bf16(a, b, c);
    }
    const int d = d0 + ln;
#pragma unroll
    for (int r = 0; r < 8; ++r) {
      const int q = q0 + r + 8 * hf;
      Xctx[((size_t)bb * S_ + q) * D_ + h * DH_ + d] = (__bf16)c[r];
    }
  }
}

// ---------------- Output projection: out = Xctx @ Wo^T + bo (fp32 out) ------
__global__ void __launch_bounds__(256) out_proj_kernel(
    const __bf16* __restrict__ X, const __bf16* __restrict__ W,
    const float* __restrict__ bias, float* __restrict__ out) {
  const int wave = threadIdx.x >> 5, lane = threadIdx.x & 31;
  const int wtile = blockIdx.x * 8 + wave;
  const int m0 = (wtile >> 4) * 64;
  const int n0 = (wtile & 15) * 64;
  v8f acc[4][4];
#pragma unroll
  for (int i = 0; i < 4; ++i)
#pragma unroll
    for (int j = 0; j < 4; ++j) acc[i][j] = zero8();

  for (int k0 = 0; k0 < D_; k0 += 32) {
    v16bf a[4], b[4];
#pragma unroll
    for (int i = 0; i < 4; ++i) a[i] = frag_a_bf16(X, D_, m0 + 16 * i, k0, lane);
#pragma unroll
    for (int j = 0; j < 4; ++j) b[j] = frag_bt_bf16(W, D_, n0 + 16 * j, k0, lane);
#pragma unroll
    for (int i = 0; i < 4; ++i)
#pragma unroll
      for (int j = 0; j < 4; ++j) acc[i][j] = wmma_bf16(a[i], b[j], acc[i][j]);
  }

  const int ln = lane & 15, hf = lane >> 4;
#pragma unroll
  for (int j = 0; j < 4; ++j) {
    const int n = n0 + 16 * j + ln;
    const float bv = bias[n];
#pragma unroll
    for (int i = 0; i < 4; ++i) {
#pragma unroll
      for (int r = 0; r < 8; ++r) {
        const int m = m0 + 16 * i + r + 8 * hf;
        out[(size_t)m * D_ + n] = acc[i][j][r] + bv;
      }
    }
  }
}

extern "C" void kernel_launch(void* const* d_in, const int* in_sizes, int n_in,
                              void* d_out, int out_size, void* d_ws, size_t ws_size,
                              hipStream_t stream) {
  const float* query = (const float*)d_in[0];
  const float* key   = (const float*)d_in[1];
  const float* value = (const float*)d_in[2];
  const int*   mask  = (const int*)d_in[3];
  const float* Wq = (const float*)d_in[4];  const float* bq = (const float*)d_in[5];
  const float* Wk = (const float*)d_in[6];  const float* bk = (const float*)d_in[7];
  const float* Wv = (const float*)d_in[8];  const float* bv = (const float*)d_in[9];
  const float* Wo = (const float*)d_in[10]; const float* bo = (const float*)d_in[11];

  float* out  = (float*)d_out;
  float* attn = out + (size_t)B_ * S_ * D_;   // attention follows `out` in d_out

  const size_t nBSD = (size_t)B_ * S_ * D_;   // 4,194,304 elements
  const size_t nDD  = (size_t)D_ * D_;        // 1,048,576 elements
  char* ws = (char*)d_ws;
  size_t off = 0;
  __bf16* Qbf  = (__bf16*)(ws + off); off += nBSD * 2;   // 8 MB  [b,h,s,dh]
  __bf16* Kbf  = (__bf16*)(ws + off); off += nBSD * 2;   // 8 MB  [b,h,s,dh]
  __bf16* Vt   = (__bf16*)(ws + off); off += nBSD * 2;   // 8 MB  [b,h,dh,s]
  __bf16* Xctx = (__bf16*)(ws + off); off += nBSD * 2;   // 8 MB  [m, n]
  __bf16* Xq   = (__bf16*)(ws + off); off += nBSD * 2;   // 8 MB  query bf16
  __bf16* Xk   = (__bf16*)(ws + off); off += nBSD * 2;   // 8 MB  key bf16
  __bf16* Xv   = (__bf16*)(ws + off); off += nBSD * 2;   // 8 MB  value bf16
  __bf16* Wqb  = (__bf16*)(ws + off); off += nDD * 2;    // 2 MB
  __bf16* Wkb  = (__bf16*)(ws + off); off += nDD * 2;    // 2 MB
  __bf16* Wvb  = (__bf16*)(ws + off); off += nDD * 2;    // 2 MB
  __bf16* Wob  = (__bf16*)(ws + off); off += nDD * 2;    // 2 MB

  // One-shot fp32 -> bf16 staging (keeps all converts out of GEMM hot loops).
  cvt_bf16_kernel<<<1024, 256, 0, stream>>>(query, Xq, (int)nBSD);
  cvt_bf16_kernel<<<1024, 256, 0, stream>>>(key,   Xk, (int)nBSD);
  cvt_bf16_kernel<<<1024, 256, 0, stream>>>(value, Xv, (int)nBSD);
  cvt_bf16_kernel<<<256, 256, 0, stream>>>(Wq, Wqb, (int)nDD);
  cvt_bf16_kernel<<<256, 256, 0, stream>>>(Wk, Wkb, (int)nDD);
  cvt_bf16_kernel<<<256, 256, 0, stream>>>(Wv, Wvb, (int)nDD);
  cvt_bf16_kernel<<<256, 256, 0, stream>>>(Wo, Wob, (int)nDD);

  // 1024 wave-blocks (64x64 each) per projection; 8 waves / 256-thread block.
  proj_kernel<0><<<128, 256, 0, stream>>>(Xq, Wqb, bq, Qbf);
  proj_kernel<0><<<128, 256, 0, stream>>>(Xk, Wkb, bk, Kbf);
  proj_kernel<1><<<128, 256, 0, stream>>>(Xv, Wvb, bv, Vt);

  // One block per (b,h, 16-query strip): 4096 blocks, 256 KB dynamic LDS
  // (128 KB fp32 scores + 128 KB TDM mask strip).
  attn_fused_kernel<<<4096, 256, 2 * kStripBytes, stream>>>(
      Qbf, Kbf, Vt, mask, attn, Xctx);

  out_proj_kernel<<<128, 256, 0, stream>>>(Xctx, Wob, bo, out);
}
